// ComiRecDynamicRoutingLayer_26508538150975
// MI455X (gfx1250) — matrix operand order repaired and checked by reading, not compiled, not run on hardware
//
#include <hip/hip_runtime.h>
#include <stdint.h>

// ---------------------------------------------------------------------------
// ComiRec dynamic routing for MI455X (gfx1250, wave32, WMMA).
//
// Phase 1: u[b,n,s,c] = sum_e behaviors[b,s,e] * W[n,s,e,c]
//          800 GEMMs of (1024x128)@(128x128), bf16 inputs, f32 accumulate,
//          v_wmma_f32_16x16x32_bf16. u streamed to d_ws (419 MB) with
//          non-temporal stores so W (52 MB) + behaviors (105 MB) stay
//          resident in the 192 MB L2 across their 8x / 4x reuse.
// Phase 2: per-(b,n) workgroup DMAs u[b,n] (100 KB) into LDS with
//          global_load_async_to_lds_b128 (ASYNCcnt) and runs all 3 routing
//          rounds locally -> u crosses HBM exactly once (~18 us @ 23.3 TB/s).
// ---------------------------------------------------------------------------

typedef __attribute__((ext_vector_type(16))) __bf16   v16bf;
typedef __attribute__((ext_vector_type(8)))  float    v8f;
typedef __attribute__((ext_vector_type(8)))  unsigned v8u;

#define CAPS    4
#define S_DIM   200
#define E_DIM   128
#define C_DIM   128
#define ROUNDS  3

// LDS row pitch in dwords for bf16x2-packed tiles: 64 data dwords + 1 pad
// (pad keeps the 16 lanes of a half-wave on distinct banks during frag loads)
#define LDA 65
#define LDB 65

__device__ __forceinline__ unsigned short f32_bf16(float f) {
    // round-to-nearest-even truncation f32 -> bf16
    unsigned u = __float_as_uint(f);
    u += 0x7FFFu + ((u >> 16) & 1u);
    return (unsigned short)(u >> 16);
}
__device__ __forceinline__ unsigned pack_bf16x2(float lo, float hi) {
    return (unsigned)f32_bf16(lo) | ((unsigned)f32_bf16(hi) << 16);
}

// ---------------------------------------------------------------------------
// Phase 1: u = einsum('bse,nsec->bnsc')  as 800 GEMMs, tiled 128(M)x128(C).
// grid = (B/128, S, N), block = 256 threads = 8 waves.
// Wave w computes the 16-row strip [16w,16w+16) x all 128 C columns
// (8 accumulator tiles, 4 K-steps of 32 -> 32 WMMAs per wave).
// ---------------------------------------------------------------------------
__global__ __launch_bounds__(256)
void comirec_u_gemm(const float* __restrict__ behaviors,
                    const float* __restrict__ W,
                    float* __restrict__ U) {
    __shared__ unsigned Als[128 * LDA];  // behaviors tile, [m][kd] bf16x2
    __shared__ unsigned Bls[128 * LDB];  // W tile transposed, [c][kd] bf16x2

    const int tid  = threadIdx.x;
    const int mblk = blockIdx.x;             // batch block (128 rows)
    const int s    = blockIdx.y;             // sequence position
    const int n    = blockIdx.z;             // capsule
    const int b0   = mblk * 128;

    // ---- stage A = behaviors[b0..b0+127][s][0..127] as bf16 -----------------
    {
        const float* src = behaviors + (size_t)s * E_DIM;
        #pragma unroll 4
        for (int it = 0; it < 32; ++it) {
            int i  = it * 256 + tid;         // 8192 dword elements
            int m  = i >> 6;
            int kd = i & 63;
            float2 f = *(const float2*)(src + (size_t)(b0 + m) * (S_DIM * E_DIM) + 2 * kd);
            Als[m * LDA + kd] = pack_bf16x2(f.x, f.y);
        }
    }
    // ---- stage B^T : W[n][s][k][c] -> Bls[c][k] as bf16 ---------------------
    {
        const float* src = W + ((size_t)n * S_DIM + s) * (E_DIM * C_DIM);
        unsigned short* B16 = (unsigned short*)Bls;   // row pitch = 2*LDB ushorts
        #pragma unroll 4
        for (int it = 0; it < 32; ++it) {
            int i  = it * 256 + tid;
            int k  = i >> 6;
            int c2 = i & 63;                 // float2 column index (coalesced)
            float2 f = *(const float2*)(src + (size_t)k * C_DIM + 2 * c2);
            B16[(2 * c2 + 0) * (2 * LDB) + k] = f32_bf16(f.x);
            B16[(2 * c2 + 1) * (2 * LDB) + k] = f32_bf16(f.y);
        }
    }
    __syncthreads();

    const int wave = tid >> 5;
    const int lane = tid & 31;
    const int half = lane >> 4;   // lane half selects K-subgroup per ISA layout
    const int l16  = lane & 15;
    const int m0   = wave * 16;

    v8f acc[8];
    v8f zero = {0.f, 0.f, 0.f, 0.f, 0.f, 0.f, 0.f, 0.f};
    #pragma unroll
    for (int nt = 0; nt < 8; ++nt) acc[nt] = zero;

    #pragma unroll
    for (int ks = 0; ks < 4; ++ks) {
        const int kd0 = ks * 16;  // 32 bf16 = 16 dwords per K-step

        // A fragment 16x32 (CDNA5 16-bit A layout):
        //   lanes 0-15 : M=lane,   v0-3 -> K 0-7,   v4-7 -> K 16-23
        //   lanes16-31 : M=lane-16,v0-3 -> K 8-15,  v4-7 -> K 24-31
        v8u au;
        {
            const unsigned* row = &Als[(m0 + l16) * LDA];
            const int base = kd0 + 4 * half;
            #pragma unroll
            for (int v = 0; v < 4; ++v) au[v]     = row[base + v];
            #pragma unroll
            for (int v = 0; v < 4; ++v) au[4 + v] = row[base + 8 + v];
        }
        v16bf a = __builtin_bit_cast(v16bf, au);

        #pragma unroll
        for (int nt = 0; nt < 8; ++nt) {
            // B fragment 32x16: lanes 0-15 N=lane K0-7,16-23, lanes 16-31 N K8-15,24-31
            const unsigned* row = &Bls[(nt * 16 + l16) * LDB];
            const int base = kd0 + 4 * half;
            v8u bu;
            #pragma unroll
            for (int v = 0; v < 8; ++v) bu[v] = row[base + v + (v & 4)];
            v16bf bb = __builtin_bit_cast(v16bf, bu);

            acc[nt] = __builtin_amdgcn_wmma_f32_16x16x32_bf16(
                false, a, false, bb, (short)0, acc[nt], false, false);
        }
    }

    // ---- store D tiles to u[b][n][s][c] (f32 D layout: vgpr r, lane l ->
    //      M = r + 8*(l>>4), N = l&15). Non-temporal: u is write-once and
    //      419 MB; keep it out of L2 so W/behaviors stay resident. ----------
    #pragma unroll
    for (int nt = 0; nt < 8; ++nt) {
        const int c = nt * 16 + l16;
        #pragma unroll
        for (int r = 0; r < 8; ++r) {
            const int mrow = r + 8 * half;
            const size_t b = (size_t)(b0 + m0 + mrow);
            __builtin_nontemporal_store(
                acc[nt][r], &U[((b * CAPS + n) * S_DIM + s) * C_DIM + c]);
        }
    }
}

// ---------------------------------------------------------------------------
// Phase 2 helpers: wave32 shuffle reduction + 8-wave block reduction
// ---------------------------------------------------------------------------
__device__ __forceinline__ float wave_reduce(float v, bool is_max) {
    #pragma unroll
    for (int off = 16; off >= 1; off >>= 1) {
        float o = __shfl_xor(v, off, 32);
        v = is_max ? fmaxf(v, o) : (v + o);
    }
    return v;
}

__device__ __forceinline__ float block_reduce_256(float v, bool is_max, int tid,
                                                  float* red, float* redOut) {
    v = wave_reduce(v, is_max);
    if ((tid & 31) == 0) red[tid >> 5] = v;
    __syncthreads();
    if (tid == 0) {
        float r = red[0];
        #pragma unroll
        for (int w = 1; w < 8; ++w) r = is_max ? fmaxf(r, red[w]) : (r + red[w]);
        *redOut = r;
    }
    __syncthreads();
    float r = *redOut;
    __syncthreads();           // allow red/redOut reuse in the next reduction
    return r;
}

// ---------------------------------------------------------------------------
// Phase 2: dynamic routing, one workgroup per (b,n). u[b,n] staged in LDS
// via CDNA5 async DMA, so u crosses HBM exactly once. grid = B*CAPS.
// ---------------------------------------------------------------------------
__global__ __launch_bounds__(256)
void comirec_routing(const float* __restrict__ U,
                     const int*   __restrict__ valid_mask,
                     float*       __restrict__ out) {
    __shared__ __align__(16) float Uls[S_DIM * C_DIM];   // 100 KB
    __shared__ float logits[S_DIM];
    __shared__ float coup[S_DIM];
    __shared__ float capsv[C_DIM];
    __shared__ float red[8];
    __shared__ float redOut;
    __shared__ int   maskl[S_DIM];

    const int tid = threadIdx.x;
    const int bn  = blockIdx.x;      // b*CAPS + n
    const int b   = bn >> 2;

    // ---- async DMA u[b,n] (25600 f32 = 6400 x B128 = 25 iters x 256 thr)
    //      straight into LDS: no VGPR round-trip, tracked with ASYNCcnt.
    //      LDS address = low 32 bits of the generic shared-space pointer. ----
    {
        const uint64_t gbase = (uint64_t)(uintptr_t)(U + (size_t)bn * (S_DIM * C_DIM));
        const unsigned lbase = (unsigned)(uintptr_t)(&Uls[0]);
        #pragma unroll 5
        for (int it = 0; it < 25; ++it) {
            const int i = it * 256 + tid;          // B128 transfer index
            const uint64_t gaddr = gbase + (uint64_t)i * 16u;
            const unsigned laddr = lbase + (unsigned)i * 16u;
            asm volatile("global_load_async_to_lds_b128 %0, %1, off"
                         :: "v"(laddr), "v"(gaddr) : "memory");
        }
    }
    if (tid < S_DIM) {
        maskl[tid]  = valid_mask[b * S_DIM + tid];
        logits[tid] = 0.0f;
    }
    asm volatile("s_wait_asynccnt 0x0" ::: "memory");  // my wave's DMAs landed
    __syncthreads();                                   // everyone's DMAs landed

    for (int rd = 0; rd < ROUNDS; ++rd) {
        // ---- masked softmax over s ------------------------------------------
        const bool live = (tid < S_DIM) && (maskl[tid] != 0);
        float lv = live ? logits[tid] : -3.402823466e38f;
        float mx = block_reduce_256(lv, true, tid, red, &redOut);
        float ev = live ? __expf(logits[tid] - mx) : 0.0f;
        float sm = block_reduce_256(ev, false, tid, red, &redOut);
        if (tid < S_DIM) coup[tid] = ev / sm;
        __syncthreads();

        // ---- caps_raw[c] = sum_s coup[s] * u[s,c]  (thread == c) ------------
        if (tid < C_DIM) {
            float a = 0.0f;
            for (int s = 0; s < S_DIM; ++s)
                a = fmaf(coup[s], Uls[s * C_DIM + tid], a);   // coup: broadcast
            capsv[tid] = a;
        }
        __syncthreads();

        // ---- squash: scale = n2/((1+n2)*n) == n/(1+n2) ----------------------
        float sq = (tid < C_DIM) ? capsv[tid] * capsv[tid] : 0.0f;
        float n2 = block_reduce_256(sq, false, tid, red, &redOut);
        float scale = __fsqrt_rn(n2) / (1.0f + n2);
        if (tid < C_DIM) capsv[tid] *= scale;
        __syncthreads();

        // ---- logits[s] += caps . u[s,:]  (one wave-dot per s) ---------------
        if (rd < ROUNDS - 1) {
            const int wave = tid >> 5, lane = tid & 31;
            for (int s = wave; s < S_DIM; s += 8) {
                float p = 0.0f;
                #pragma unroll
                for (int c = lane; c < C_DIM; c += 32)
                    p = fmaf(capsv[c], Uls[s * C_DIM + c], p);
                p = wave_reduce(p, false);
                if (lane == 0) logits[s] += p;
            }
            __syncthreads();
        }
    }

    if (tid < C_DIM) out[(size_t)bn * C_DIM + tid] = capsv[tid];
}

// ---------------------------------------------------------------------------
extern "C" void kernel_launch(void* const* d_in, const int* in_sizes, int n_in,
                              void* d_out, int out_size, void* d_ws, size_t ws_size,
                              hipStream_t stream) {
    const float* behaviors = (const float*)d_in[0];            // (B,S,E) f32
    const int*   valid     = (const int*)  d_in[1];            // (B,S)
    const float* W         = (const float*)d_in[2];            // (N,S,E,C) f32
    float* out = (float*)d_out;                                // (B,N,C) f32
    float* U   = (float*)d_ws;                                 // (B,N,S,C) f32 scratch

    const int B = in_sizes[0] / (S_DIM * E_DIM);               // 1024

    dim3 g1(B / 128, S_DIM, CAPS);                             // 6400 workgroups
    comirec_u_gemm<<<g1, 256, 0, stream>>>(behaviors, W, U);

    dim3 g2(B * CAPS);                                         // 4096 workgroups
    comirec_routing<<<g2, 256, 0, stream>>>(U, valid, out);
}
